// IDAUp_53171695125007
// MI455X (gfx1250) — matrix operand order repaired and checked by reading, not compiled, not run on hardware
//
#include <hip/hip_runtime.h>
#include <hip/hip_bf16.h>
#include <cstdint>

typedef _Float16 v16h __attribute__((ext_vector_type(16)));
typedef _Float16 v8h  __attribute__((ext_vector_type(8)));
typedef float    v8f  __attribute__((ext_vector_type(8)));
typedef unsigned int v4u __attribute__((ext_vector_type(4)));
typedef int      v4i __attribute__((ext_vector_type(4)));
typedef int      v8i __attribute__((ext_vector_type(8)));

#define OCH 64
#define BEPS 1e-5f

// ---------------- NCHW -> NHWC transpose, optional fused add of 2nd NCHW ----------------
__global__ void k_to_nhwc(const float* __restrict__ a, const float* __restrict__ b2,
                          float* __restrict__ out, int C, int H, int W, int n) {
  int idx = blockIdx.x * 256 + threadIdx.x;
  if (idx >= n) return;
  int c = idx % C; int t = idx / C;
  int w = t % W; t /= W;
  int h = t % H; int b = t / H;
  size_t src = (((size_t)b * C + c) * H + h) * W + w;
  float v = a[src];
  if (b2) v += b2[src];
  out[idx] = v;
}

// ------- main-conv weight prep: w(64,C,3,3) f32 -> wt[kk][chunk][n(64)][c(64)] f16 -------
__global__ void k_wprep(const float* __restrict__ w, _Float16* __restrict__ wt, int C, int n) {
  int idx = blockIdx.x * 256 + threadIdx.x;
  if (idx >= n) return;
  int c   = idx & 63;
  int nch = (idx >> 6) & 63;
  int t   = idx >> 12;
  int nchunk = C >> 6;
  int ch = t % nchunk; int kk = t / nchunk;
  wt[idx] = (_Float16)w[((size_t)nch * C + (ch * 64 + c)) * 9 + kk];
}

// ------- offset-conv weight prep: w(27,C,3,3) f32 -> wto[kk][chunk][n(32,pad0)][c(64)] f16 -------
__global__ void k_wprep_off(const float* __restrict__ w, _Float16* __restrict__ wto, int C, int n) {
  int idx = blockIdx.x * 256 + threadIdx.x;
  if (idx >= n) return;
  int c   = idx & 63;
  int nch = (idx >> 6) & 31;
  int t   = idx >> 11;
  int nchunk = C >> 6;
  int ch = t % nchunk; int kk = t / nchunk;
  wto[idx] = (nch < 27) ? (_Float16)w[((size_t)nch * C + (ch * 64 + c)) * 9 + kk]
                        : (_Float16)0.f;
}

// ---------------- depthwise transposed conv (lhs_dilation=s), NCHW -> NCHW ----------------
__global__ void k_upconv(const float* __restrict__ in, const float* __restrict__ w,
                         float* __restrict__ out, int C, int Hin, int Win, int s, int n) {
  int idx = blockIdx.x * 256 + threadIdx.x;
  if (idx >= n) return;
  int K = 2 * s; int pb = 2 * s - 1 - s / 2;
  int Wo = Win * s, Ho = Hin * s;
  int ox = idx % Wo; int t = idx / Wo;
  int oy = t % Ho; t /= Ho;
  int c = t % C; int b = t / C;
  float acc = 0.f;
  for (int ky = 0; ky < K; ++ky) {
    int yy = oy + ky - pb;
    if (yy < 0 || (yy % s) != 0) continue;
    int iy = yy / s; if (iy >= Hin) continue;
    for (int kx = 0; kx < K; ++kx) {
      int xx = ox + kx - pb;
      if (xx < 0 || (xx % s) != 0) continue;
      int ix = xx / s; if (ix >= Win) continue;
      acc += w[((size_t)c * K + (K - 1 - ky)) * K + (K - 1 - kx)] *
             in[(((size_t)b * C + c) * Hin + iy) * Win + ix];
    }
  }
  out[idx] = acc;
}

// ---------------- TDM: load a (rows x 64) f16 tile into LDS with 8-f16 row padding ----------------
// D# per ISA 8.3/8.4: data_size=1 (2B), pad_enable, pad_interval=4 (32 DWORDs = 128B row),
// pad_amount=3 (4 DWORDs = 8 f16) -> 144B LDS row pitch, matching [rows][72] tiles.
__device__ __forceinline__ void tdm_load_tile_f16(void* lds_dst, const void* gsrc, int rows) {
  unsigned lds_off = (unsigned)(unsigned long long)(uintptr_t)lds_dst; // low 32b of flat = LDS offset
  unsigned long long ga = (unsigned long long)(uintptr_t)gsrc;
  v4u g0;
  g0[0] = 1u;                                            // count=1 (valid user D#)
  g0[1] = lds_off;                                       // lds_addr
  g0[2] = (unsigned)(ga & 0xFFFFFFFFu);                  // global_addr lo
  g0[3] = (unsigned)((ga >> 32) & 0x1FFFFFFu) | (2u << 30); // global_addr hi | type=2
  v8i g1;
  g1[0] = (int)((1u << 16) | (1u << 20) | (4u << 22) | (3u << 25));
  g1[1] = (64 << 16);        // tensor_dim0 = 64
  g1[2] = (rows << 16);      // tensor_dim1 = rows
  g1[3] = (64 << 16);        // tile_dim0 = 64
  g1[4] = rows;              // tile_dim1 = rows, tile_dim2 = 0
  g1[5] = 64;                // tensor_dim0_stride = 64
  g1[6] = 0;
  g1[7] = 0;
  v4i g2 = {0, 0, 0, 0};
  v4i g3 = {0, 0, 0, 0};
#if defined(__clang_major__) && (__clang_major__ >= 23)
  v8i g4 = {0, 0, 0, 0, 0, 0, 0, 0};
  __builtin_amdgcn_tensor_load_to_lds(g0, g1, g2, g3, g4, 0);
#else
  __builtin_amdgcn_tensor_load_to_lds(g0, g1, g2, g3, 0);
#endif
}

// ---------------- WMMA fragment loaders (A 16x32 / B 32x16, contiguous k = 16*(lane>>4) + s) ----------------
__device__ __forceinline__ v16h load_afrag(const _Float16* arow, int ksOff, int kbA) {
  v8h lo = *(const v8h*)(arow + ksOff + kbA);
  v8h hi = *(const v8h*)(arow + ksOff + kbA + 8);
  return __builtin_shufflevector(lo, hi, 0,1,2,3,4,5,6,7,8,9,10,11,12,13,14,15);
}
__device__ __forceinline__ v16h load_bfrag(const _Float16* brow, int ksOff, int kbB) {
  v8h lo = *(const v8h*)(brow + ksOff + kbB);
  v8h hi = *(const v8h*)(brow + ksOff + kbB + 8);
  return __builtin_shufflevector(lo, hi, 0,1,2,3,4,5,6,7,8,9,10,11,12,13,14,15);
}

// ---------------- offset conv via WMMA (im2col, N=27 padded to 32) + sigmoid on mask ----------------
__launch_bounds__(128)
__global__ void k_offconv_wmma(const float* __restrict__ x,       // NHWC
                               const _Float16* __restrict__ wto,  // [9][C/64][32][64]
                               const float* __restrict__ bb,      // (27)
                               float* __restrict__ off,           // (npix, 27)
                               int C, int H, int W) {
  __shared__ __align__(16) _Float16 Alds[4][16][72];
  __shared__ __align__(16) _Float16 Blds[32][72];

  const int tid  = threadIdx.x;
  const int wv   = tid >> 5;
  const int lane = tid & 31;
  const int tileBase = blockIdx.x * 64 + wv * 16;
  const int nchunk = C >> 6;

  v8f acc0 = {}, acc1 = {};

  for (int kk = 0; kk < 9; ++kk) {
    const int ky = kk / 3 - 1, kx = kk % 3 - 1;
    for (int c0 = 0; c0 < C; c0 += 64) {
      __syncthreads();
      if (wv == 0)
        tdm_load_tile_f16(&Blds[0][0], wto + (((size_t)kk * nchunk + (c0 >> 6)) << 11), 32);
      // A: im2col rows (zero outside image)
      for (int m = 0; m < 16; ++m) {
        int pix = tileBase + m;
        int wloc = pix % W; int t = pix / W;
        int h = t % H; int b = t / H;
        int yy = h + ky, xx = wloc + kx;
        bool valid = (yy >= 0) && (yy < H) && (xx >= 0) && (xx < W);
        const float* r = x + (((size_t)b * H + yy) * W + xx) * C + c0;
        #pragma unroll
        for (int half = 0; half < 2; ++half) {
          int cl = lane + half * 32;
          float v = valid ? r[cl] : 0.f;
          Alds[wv][m][cl] = (_Float16)v;
        }
      }
      if (wv == 0) __builtin_amdgcn_s_wait_tensorcnt(0);
      __syncthreads();
      const int mrow = lane & 15;
      const int kbA  = (lane >> 4) * 16;
      const int kbB  = (lane >> 4) * 16;
      const int nloc = lane & 15;
      #pragma unroll
      for (int ks = 0; ks < 2; ++ks) {
        const int ksOff = ks * 32;
        v16h afrag = load_afrag(&Alds[wv][mrow][0], ksOff, kbA);
        v16h b0 = load_bfrag(&Blds[ 0 + nloc][0], ksOff, kbB);
        acc0 = __builtin_amdgcn_wmma_f32_16x16x32_f16(false, afrag, false, b0, (short)0, acc0, false, false);
        v16h b1 = load_bfrag(&Blds[16 + nloc][0], ksOff, kbB);
        acc1 = __builtin_amdgcn_wmma_f32_16x16x32_f16(false, afrag, false, b1, (short)0, acc1, false, false);
      }
    }
  }

  const int nloc = lane & 15;
  const int mhi  = (lane >> 4) * 8;
  #pragma unroll
  for (int jt = 0; jt < 2; ++jt) {
    v8f a = (jt == 0) ? acc0 : acc1;
    int n = jt * 16 + nloc;
    if (n < 27) {
      float bi = bb[n];
      #pragma unroll
      for (int r = 0; r < 8; ++r) {
        int pix = tileBase + mhi + r;
        float v = a[r] + bi;
        if (n >= 18) v = 1.0f / (1.0f + __expf(-v));
        off[(size_t)pix * 27 + n] = v;
      }
    }
  }
}

// ---------------- DCN main: bilinear gather -> f16 LDS tiles -> WMMA -> BN+ReLU ----------------
__launch_bounds__(128)
__global__ void k_dcn_wmma(const float* __restrict__ x,       // NHWC (B,H,W,C)
                           const float* __restrict__ off,     // (B*H*W, 27)
                           const _Float16* __restrict__ wt,   // [9][C/64][64][64] f16
                           const float* __restrict__ bias,
                           const float* __restrict__ bng, const float* __restrict__ bnb,
                           const float* __restrict__ bnm, const float* __restrict__ bnv,
                           float* __restrict__ out,           // NCHW (B,64,H,W)
                           int C, int H, int W) {
  __shared__ __align__(16) _Float16 Alds[4][16][72];
  __shared__ __align__(16) _Float16 Blds[64][72];

  const int tid  = threadIdx.x;
  const int wv   = tid >> 5;
  const int lane = tid & 31;
  const int tileBase = blockIdx.x * 64 + wv * 16;
  const int HW = H * W;
  const int nchunk = C >> 6;

  v8f acc0 = {}, acc1 = {}, acc2 = {}, acc3 = {};

  for (int kk = 0; kk < 9; ++kk) {
    const float fky = (float)(kk / 3 - 1);
    const float fkx = (float)(kk % 3 - 1);
    for (int c0 = 0; c0 < C; c0 += 64) {
      __syncthreads();   // previous iteration's fragment loads done
      if (wv == 0)
        tdm_load_tile_f16(&Blds[0][0], wt + (((size_t)kk * nchunk + (c0 >> 6)) << 12), 64);
      // A: bilinear-gathered, mask-modulated samples
      for (int m = 0; m < 16; ++m) {
        int pix = tileBase + m;
        int wloc = pix % W; int t = pix / W;
        int h = t % H; int b = t / H;
        const float* op = off + (size_t)pix * 27;
        float dy = op[kk], dx = op[9 + kk], msk = op[18 + kk];
        float py = (float)h + fky + dy;
        float px = (float)wloc + fkx + dx;
        float y0f = floorf(py), x0f = floorf(px);
        float wy = py - y0f, wx = px - x0f;
        int y0 = (int)y0f, x0 = (int)x0f;
        float w00 = (1.f - wy) * (1.f - wx), w01 = (1.f - wy) * wx;
        float w10 = wy * (1.f - wx),         w11 = wy * wx;
        bool v0y = (y0 >= 0) && (y0 < H);
        bool v1y = (y0 + 1 >= 0) && (y0 + 1 < H);
        bool v0x = (x0 >= 0) && (x0 < W);
        bool v1x = (x0 + 1 >= 0) && (x0 + 1 < W);
        const float* base = x + (size_t)b * HW * C;
        const float* r00 = base + ((ptrdiff_t)y0 * W + x0) * (ptrdiff_t)C;
        #pragma unroll
        for (int half = 0; half < 2; ++half) {
          int c = c0 + lane + half * 32;
          float v = 0.f;
          if (v0y && v0x) v += w00 * r00[c];
          if (v0y && v1x) v += w01 * r00[c + C];
          if (v1y && v0x) v += w10 * r00[c + (ptrdiff_t)W * C];
          if (v1y && v1x) v += w11 * r00[c + (ptrdiff_t)W * C + C];
          Alds[wv][m][lane + half * 32] = (_Float16)(v * msk);
        }
      }
      if (wv == 0) __builtin_amdgcn_s_wait_tensorcnt(0);
      __syncthreads();
      const int mrow = lane & 15;
      const int kbA  = (lane >> 4) * 16;
      const int kbB  = (lane >> 4) * 16;
      const int nloc = lane & 15;
      #pragma unroll
      for (int ks = 0; ks < 2; ++ks) {
        const int ksOff = ks * 32;
        v16h afrag = load_afrag(&Alds[wv][mrow][0], ksOff, kbA);
        v16h b0 = load_bfrag(&Blds[ 0 + nloc][0], ksOff, kbB);
        acc0 = __builtin_amdgcn_wmma_f32_16x16x32_f16(false, afrag, false, b0, (short)0, acc0, false, false);
        v16h b1 = load_bfrag(&Blds[16 + nloc][0], ksOff, kbB);
        acc1 = __builtin_amdgcn_wmma_f32_16x16x32_f16(false, afrag, false, b1, (short)0, acc1, false, false);
        v16h b2 = load_bfrag(&Blds[32 + nloc][0], ksOff, kbB);
        acc2 = __builtin_amdgcn_wmma_f32_16x16x32_f16(false, afrag, false, b2, (short)0, acc2, false, false);
        v16h b3 = load_bfrag(&Blds[48 + nloc][0], ksOff, kbB);
        acc3 = __builtin_amdgcn_wmma_f32_16x16x32_f16(false, afrag, false, b3, (short)0, acc3, false, false);
      }
    }
  }

  // epilogue: +bias, BN, ReLU, store NCHW.  C/D: VGPR r -> M = 8*(lane>>4)+r, N = lane&15.
  const int nloc = lane & 15;
  const int mhi  = (lane >> 4) * 8;
  #pragma unroll
  for (int jt = 0; jt < 4; ++jt) {
    v8f a = (jt == 0) ? acc0 : (jt == 1) ? acc1 : (jt == 2) ? acc2 : acc3;
    int n = jt * 16 + nloc;
    float inv = bng[n] * rsqrtf(bnv[n] + BEPS);
    float sh  = bnb[n] - bnm[n] * inv;
    float bi  = bias[n];
    #pragma unroll
    for (int r = 0; r < 8; ++r) {
      int pix = tileBase + mhi + r;
      int wloc = pix % W; int t = pix / W;
      int h = t % H; int b = t / H;
      float v = (a[r] + bi) * inv + sh;
      v = fmaxf(v, 0.f);
      out[(((size_t)b * OCH + n) * H + h) * W + wloc] = v;
    }
  }
}

extern "C" void kernel_launch(void* const* d_in, const int* in_sizes, int n_in,
                              void* d_out, int out_size, void* d_ws, size_t ws_size,
                              hipStream_t stream) {
  (void)in_sizes; (void)n_in; (void)out_size; (void)ws_size;
  const float* layer0 = (const float*)d_in[0];
  const float* layer1 = (const float*)d_in[1];
  const float* layer2 = (const float*)d_in[2];

  float* ws = (float*)d_ws;
  const size_t SZ_BIG = (size_t)4 * 64 * 128 * 128;      // 4,194,304 floats
  const size_t SZ_OFF = (size_t)4 * 128 * 128 * 27;      // 1,769,472 floats
  const size_t SZ_WT  = (size_t)9 * 4 * 64 * 64;         // 147,456 halves (C up to 256)
  float* bufA   = ws;
  float* bufB   = bufA + SZ_BIG;
  float* bufC   = bufB + SZ_BIG;
  float* bufOff = bufC + SZ_BIG;
  _Float16* bufWt    = (_Float16*)(bufOff + SZ_OFF);
  _Float16* bufWtOff = bufWt + SZ_WT;

  auto to_nhwc = [&](const float* a, const float* b2, float* o, int C, int H, int W) {
    int n = 4 * C * H * W;
    k_to_nhwc<<<(n + 255) / 256, 256, 0, stream>>>(a, b2, o, C, H, W, n);
  };
  auto dcn = [&](const float* xnhwc, int p, float* o, int C, int H, int W) {
    const float* offw = (const float*)d_in[p + 0];
    const float* offb = (const float*)d_in[p + 1];
    const float* w    = (const float*)d_in[p + 2];
    const float* b    = (const float*)d_in[p + 3];
    const float* g    = (const float*)d_in[p + 4];
    const float* bb   = (const float*)d_in[p + 5];
    const float* m    = (const float*)d_in[p + 6];
    const float* v    = (const float*)d_in[p + 7];
    int nwt  = 9 * (C >> 6) * 64 * 64;
    int nwto = 9 * (C >> 6) * 32 * 64;
    k_wprep<<<(nwt + 255) / 256, 256, 0, stream>>>(w, bufWt, C, nwt);
    k_wprep_off<<<(nwto + 255) / 256, 256, 0, stream>>>(offw, bufWtOff, C, nwto);
    int npix = 4 * H * W;
    k_offconv_wmma<<<npix / 64, 128, 0, stream>>>(xnhwc, bufWtOff, offb, bufOff, C, H, W);
    k_dcn_wmma<<<npix / 64, 128, 0, stream>>>(xnhwc, bufOff, bufWt, b, g, bb, m, v, o, C, H, W);
  };
  auto up = [&](const float* in, const float* w, float* o, int Hin, int Win, int s) {
    int n = 4 * 64 * (Hin * s) * (Win * s);
    k_upconv<<<(n + 255) / 256, 256, 0, stream>>>(in, w, o, 64, Hin, Win, s, n);
  };

  // p1: DCN(layer1) -> up x2 -> +layer0 -> n1 -> l1
  to_nhwc(layer1, nullptr, bufA, 128, 64, 64);
  dcn(bufA, 3, bufB, 128, 64, 64);                       // p1 out (NCHW 64ch 64x64)
  up(bufB, (const float*)d_in[11], bufC, 64, 64, 2);     // up1 (128x128)
  to_nhwc(bufC, layer0, bufA, 64, 128, 128);             // x for n1
  dcn(bufA, 12, bufB, 64, 128, 128);                     // l1 (NCHW)

  // p2: DCN(layer2) -> up x4 -> +l1 -> n2 -> output
  to_nhwc(layer2, nullptr, bufC, 256, 32, 32);
  dcn(bufC, 20, bufA, 256, 32, 32);                      // p2 out
  up(bufA, (const float*)d_in[28], bufC, 32, 32, 4);     // up2 (128x128)
  to_nhwc(bufC, bufB, bufA, 64, 128, 128);               // x for n2 (up2 + l1)
  dcn(bufA, 29, (float*)d_out, 64, 128, 128);            // l2 -> d_out (NCHW f32)
}
